// MultiheadAttention_24833500905563
// MI455X (gfx1250) — matrix operand order repaired and checked by reading, not compiled, not run on hardware
//
#include <hip/hip_runtime.h>
#include <hip/hip_bf16.h>

// Problem constants (fixed by the reference).
#define BB 2
#define TT 2048
#define DD 768
#define HH 12
#define DHH 64
#define MM (BB * TT)            // 4096 rows in the projection GEMMs
#define SOFTMAX_SCALE 0.125f    // 1/sqrt(64)

typedef _Float16 half_t;
typedef __attribute__((ext_vector_type(16))) _Float16 v16h;
typedef __attribute__((ext_vector_type(8)))  float    v8f;

// ---------------------------------------------------------------------------
// WMMA register layout helpers (CDNA5 ISA §7.12.2, wave32)
// A-matrix (16-bit, 16x32): lane holds row M = lane&15. Half j (0..15) of the
// v16h maps to K = 16*(v>=4) + 8*(lane>=16) + 2*(v&3) + h, with v=j/2, h=j&1.
// B-matrix (16-bit, 32x16): lane&15 = N column, half j = K in the same order as A.
// C/D (f32 16x16): lane -> N = lane&15, vgpr r -> M = r + 8*(lane>=16).
// ---------------------------------------------------------------------------
__device__ __forceinline__ int a_k_of(int lane, int j) {
  int v = j >> 1, hb = j & 1;
  return ((v & 4) ? 16 : 0) + ((lane & 16) ? 8 : 0) + ((v & 3) << 1) + hb;
}

__device__ __forceinline__ v8f wmma_f16(v16h a, v16h b, v8f c) {
  return __builtin_amdgcn_wmma_f32_16x16x32_f16(
      /*neg_a=*/false, a, /*neg_b=*/false, b,
      /*c_mod=*/(short)0, c, /*reuse_a=*/false, /*reuse_b=*/false);
}

// ---------------------------------------------------------------------------
// fp32 -> f16 conversion (one-time, ~16 MB total)
// ---------------------------------------------------------------------------
__global__ __launch_bounds__(256) void cvt_f32_f16(const float* __restrict__ s,
                                                   half_t* __restrict__ d, int n) {
  int i = blockIdx.x * 256 + threadIdx.x;
  if (i < n) d[i] = (half_t)s[i];
}

// ---------------------------------------------------------------------------
// WMMA GEMM: out = A[MM x DD] * W[DD x DD] + bias, one wave per 16x64 strip.
// mode 0: scatter f16 output into head-major [B,H,T,DH]      (Q, V)
// mode 1: fp32 row-major output (final projection into d_out)
// mode 2: scatter f16 output into dh-major  [B,H,DH,T]       (K transposed)
// ---------------------------------------------------------------------------
__global__ __launch_bounds__(32)
void gemm16x64_wmma(const half_t* __restrict__ A, const half_t* __restrict__ W,
                    const float* __restrict__ bias, half_t* __restrict__ out16,
                    float* __restrict__ out32, int mode) {
  const int lane = threadIdx.x;
  const int m0 = blockIdx.x * 16;
  const int n0 = blockIdx.y * 64;

  v8f acc[4] = {};
  const half_t* arow = A + (size_t)(m0 + (lane & 15)) * DD;

  for (int kc = 0; kc < DD; kc += 32) {
    // A tile (16x32) in A-layout (halves coalesce into 2x b128 per lane)
    v16h a;
#pragma unroll
    for (int j = 0; j < 16; ++j) a[j] = arow[kc + a_k_of(lane, j)];

    // B tiles: lane&15 = N column, halves = K in the same order as A
    const half_t* wb = W + (size_t)kc * DD + n0 + (lane & 15);
#pragma unroll
    for (int nt = 0; nt < 4; ++nt) {
      v16h b;
#pragma unroll
      for (int j = 0; j < 16; ++j) b[j] = wb[(size_t)a_k_of(lane, j) * DD + nt * 16];
      acc[nt] = wmma_f16(a, b, acc[nt]);
    }
  }

  const int nn  = lane & 15;
  const int mhi = (lane & 16) ? 8 : 0;
#pragma unroll
  for (int nt = 0; nt < 4; ++nt) {
#pragma unroll
    for (int r = 0; r < 8; ++r) {
      int row = m0 + mhi + r;
      int col = n0 + nt * 16 + nn;
      float val = acc[nt][r] + bias[col];
      if (mode == 1) {
        out32[(size_t)row * DD + col] = val;
      } else {
        int b = row / TT, t = row - b * TT;
        int h = col / DHH, dh = col - h * DHH;
        if (mode == 0) {
          out16[(((size_t)b * HH + h) * TT + t) * DHH + dh] = (half_t)val;
        } else {  // mode 2: K transposed, dh-major
          out16[(((size_t)b * HH + h) * DHH + dh) * TT + t] = (half_t)val;
        }
      }
    }
  }
}

// ---------------------------------------------------------------------------
// Flash attention: one wave per (b, h, 16-row q tile). Streams keys in
// 32-wide steps; scores stay in registers/LDS (never HBM).
// K is dh-major so score B-tiles are contiguous 32B vector loads.
// ---------------------------------------------------------------------------
__global__ __launch_bounds__(32)
void attn_flash_wmma(const half_t* __restrict__ q16, const half_t* __restrict__ kT16,
                     const half_t* __restrict__ v16p, half_t* __restrict__ ctx16) {
  const int lane = threadIdx.x;
  const int q0 = blockIdx.x * 16;
  const int h  = blockIdx.y;
  const int b  = blockIdx.z;
  const size_t hoff = ((size_t)b * HH + h) * TT * DHH;
  const half_t* Q   = q16  + hoff;
  const half_t* KT  = kT16 + hoff;   // [DH][T] within head
  const half_t* Vh  = v16p + hoff;   // [T][DH] within head

  // Padded LDS (odd strides -> bank-conflict free row access)
  __shared__ float  sc[16][33];      // raw score tile (D-layout dump)
  __shared__ half_t pp[16][34];      // softmax probabilities, row-major
  __shared__ float  hmax[2][16];     // half-wave max partials
  __shared__ float  hsum[2][16];     // half-wave sum partials
  __shared__ float  m_sh[16], l_sh[16], alpha_sh[16], linv_sh[16];

  const int nn  = lane & 15;
  const int mhi = (lane & 16) ? 8 : 0;
  const int hw  = lane >> 4;         // half-wave id (0/1)
  const int jb  = hw << 4;           // key sub-range base (0 or 16)

  if (lane < 16) { m_sh[lane] = -3.0e38f; l_sh[lane] = 0.0f; }

  // Q tile (16 rows x 64 dh) resident in A-layout registers for whole loop
  v16h qa0, qa1;
  {
    const half_t* qrow = Q + (size_t)(q0 + nn) * DHH;
#pragma unroll
    for (int j = 0; j < 16; ++j) {
      int k = a_k_of(lane, j);
      qa0[j] = qrow[k];
      qa1[j] = qrow[32 + k];
    }
  }

  // B-layout base pointers for K^T: lane&15 = key column, halves = dh (A order)
  const half_t* ktrow0 = KT + nn;                        // dh chunk 0 (0..31)
  const half_t* ktrow1 = KT + (size_t)32 * TT + nn;      // dh chunk 1 (32..63)

  v8f acc[4] = {};
  __builtin_amdgcn_wave_barrier();

  const int kend = q0 + 16;              // causal: keys 0 .. q0+15
  for (int kb = 0; kb < kend; kb += 32) {
    // ---- scores: two 16-key tiles, K-dim = DH split in two 32-chunks ----
#pragma unroll
    for (int kt = 0; kt < 2; ++kt) {
      const int kcol = kb + kt * 16;
      v16h b0, b1;
#pragma unroll
      for (int j = 0; j < 16; ++j) {
        size_t ko = (size_t)a_k_of(lane, j) * TT + kcol;
        b0[j] = ktrow0[ko];
        b1[j] = ktrow1[ko];
      }
      v8f st = {};
      st = wmma_f16(qa0, b0, st);
      st = wmma_f16(qa1, b1, st);
#pragma unroll
      for (int r = 0; r < 8; ++r) sc[mhi + r][kt * 16 + nn] = st[r];
    }
    __builtin_amdgcn_wave_barrier();

    // ---- online softmax, parallel across all 32 lanes ----
    // each lane owns row nn=lane&15 and 16 of the 32 keys (jb..jb+15)
    const int qg = q0 + nn;
    int jmax = qg - kb; if (jmax > 31) jmax = 31;   // causal bound (>= 0 here)

    float pmx = -3.0e38f;
#pragma unroll 4
    for (int j = 0; j < 16; ++j) {
      int jj = jb + j;
      if (jj <= jmax) pmx = fmaxf(pmx, sc[nn][jj] * SOFTMAX_SCALE);
    }
    hmax[hw][nn] = pmx;
    __builtin_amdgcn_wave_barrier();

    float mx = fmaxf(m_sh[nn], fmaxf(hmax[0][nn], hmax[1][nn]));
    float psum = 0.0f;
#pragma unroll 4
    for (int j = 0; j < 16; ++j) {
      int jj = jb + j;
      float e = (jj <= jmax) ? __expf(sc[nn][jj] * SOFTMAX_SCALE - mx) : 0.0f;
      pp[nn][jj] = (half_t)e;
      psum += e;
    }
    hsum[hw][nn] = psum;
    __builtin_amdgcn_wave_barrier();

    if (lane < 16) {
      float al = __expf(m_sh[lane] - mx);          // nn == lane here
      l_sh[lane] = l_sh[lane] * al + hsum[0][lane] + hsum[1][lane];
      m_sh[lane] = mx;
      alpha_sh[lane] = al;
    }
    __builtin_amdgcn_wave_barrier();

    // ---- ctx = ctx*alpha + P(16x32) @ V(32 x 64) ----
    v16h pa;                             // P in A-layout from LDS
#pragma unroll
    for (int j = 0; j < 16; ++j) pa[j] = pp[nn][a_k_of(lane, j)];
    float alv[8];
#pragma unroll
    for (int r = 0; r < 8; ++r) alv[r] = alpha_sh[mhi + r];

    const half_t* vbase = Vh + (size_t)kb * DHH + nn;  // lane&15 = dh column
#pragma unroll
    for (int nt = 0; nt < 4; ++nt) {
#pragma unroll
      for (int r = 0; r < 8; ++r) acc[nt][r] *= alv[r];
      v16h bv;
#pragma unroll
      for (int j = 0; j < 16; ++j) bv[j] = vbase[(size_t)a_k_of(lane, j) * DHH + nt * 16];
      acc[nt] = wmma_f16(pa, bv, acc[nt]);
    }
    __builtin_amdgcn_wave_barrier();
  }

  if (lane < 16) linv_sh[lane] = 1.0f / l_sh[lane];
  __builtin_amdgcn_wave_barrier();

  // store ctx (f16, row-major [B*T, 768]) for the output projection GEMM
#pragma unroll
  for (int nt = 0; nt < 4; ++nt) {
#pragma unroll
    for (int r = 0; r < 8; ++r) {
      int row = q0 + mhi + r;
      float val = acc[nt][r] * linv_sh[mhi + r];
      ctx16[((size_t)b * TT + row) * DD + h * DHH + nt * 16 + nn] = (half_t)val;
    }
  }
}

// ---------------------------------------------------------------------------
extern "C" void kernel_launch(void* const* d_in, const int* in_sizes, int n_in,
                              void* d_out, int out_size, void* d_ws, size_t ws_size,
                              hipStream_t stream) {
  (void)in_sizes; (void)n_in; (void)out_size; (void)ws_size;
  const float* x  = (const float*)d_in[0];
  const float* Wq = (const float*)d_in[1];
  const float* bq = (const float*)d_in[2];
  const float* Wk = (const float*)d_in[3];
  const float* bk = (const float*)d_in[4];
  const float* Wv = (const float*)d_in[5];
  const float* bv = (const float*)d_in[6];
  const float* Wo = (const float*)d_in[7];
  const float* bo = (const float*)d_in[8];

  // Workspace layout (f16 elements); total ~36.2 MB
  half_t* ws = (half_t*)d_ws;
  const size_t nXD = (size_t)MM * DD;   // 3,145,728
  const size_t nW  = (size_t)DD * DD;   //   589,824
  half_t* x16  = ws;
  half_t* wq16 = x16  + nXD;
  half_t* wk16 = wq16 + nW;
  half_t* wv16 = wk16 + nW;
  half_t* wo16 = wv16 + nW;
  half_t* q16  = wo16 + nW;
  half_t* kT16 = q16  + nXD;            // K stored dh-major [B,H,DH,T]
  half_t* v16p = kT16 + nXD;
  half_t* c16  = v16p + nXD;

  cvt_f32_f16<<<(int)((nXD + 255) / 256), 256, 0, stream>>>(x,  x16,  (int)nXD);
  cvt_f32_f16<<<(int)((nW  + 255) / 256), 256, 0, stream>>>(Wq, wq16, (int)nW);
  cvt_f32_f16<<<(int)((nW  + 255) / 256), 256, 0, stream>>>(Wk, wk16, (int)nW);
  cvt_f32_f16<<<(int)((nW  + 255) / 256), 256, 0, stream>>>(Wv, wv16, (int)nW);
  cvt_f32_f16<<<(int)((nW  + 255) / 256), 256, 0, stream>>>(Wo, wo16, (int)nW);

  dim3 gg(MM / 16, DD / 64);
  gemm16x64_wmma<<<gg, 32, 0, stream>>>(x16, wq16, bq, q16,  nullptr, 0);
  gemm16x64_wmma<<<gg, 32, 0, stream>>>(x16, wk16, bk, kT16, nullptr, 2);
  gemm16x64_wmma<<<gg, 32, 0, stream>>>(x16, wv16, bv, v16p, nullptr, 0);

  attn_flash_wmma<<<dim3(TT / 16, HH, BB), 32, 0, stream>>>(q16, kT16, v16p, c16);

  gemm16x64_wmma<<<gg, 32, 0, stream>>>(c16, wo16, bo, nullptr, (float*)d_out, 1);
}
